// PretrainGeometricAttention_13589276524819
// MI455X (gfx1250) — matrix operand order repaired and checked by reading, not compiled, not run on hardware
//
#include <hip/hip_runtime.h>

// ---------- sizes ----------
#define LL 512
#define DD 256
#define PP 128
#define HH 8
#define CELLS (LL*LL)          // 262144
#define KPAD1 1536             // FIN=1528 padded to 48*32

// ---------- workspace layout (bytes, all 256-aligned) ----------
static constexpr size_t OFF_PZN   = 0;                                  // ushort[514*514*160] bf16
static constexpr size_t SZ_PZN    = (size_t)514*514*160*2;              // 84,542,720
// transient buffers aliased inside the PZN region (dead before PZN is written)
static constexpr size_t OFF_PROJ  = 0;                                  // float[512*576]
static constexpr size_t OFF_LG    = (size_t)2*1024*1024;                // float[16*CELLS]
static constexpr size_t OFF_LS    = OFF_LG + (size_t)16*CELLS*4;        // float[4096*512]
static constexpr size_t OFF_QS    = SZ_PZN;
static constexpr size_t OFF_KS    = OFF_QS   + (size_t)512*192*4;
static constexpr size_t OFF_VPTS  = OFF_KS   + (size_t)512*192*4;
static constexpr size_t OFF_Q2    = OFF_VPTS + (size_t)512*192*4;
static constexpr size_t OFF_K2    = OFF_Q2   + 16384;
static constexpr size_t OFF_ST16  = OFF_K2   + 16384;
static constexpr size_t OFF_ALPH  = OFF_ST16 + 256;                     // float[4096*512] alpha[i][h][j]
static constexpr size_t OFF_AC136 = OFF_ALPH + (size_t)4096*512*4;
static constexpr size_t OFF_ST136 = OFF_AC136 + 1536;
static constexpr size_t OFF_BPCP  = OFF_ST136 + 1536;                   // ushort[184320]
static constexpr size_t OFF_AGGR  = OFF_BPCP + (size_t)184320*2;
static constexpr size_t OFF_AGGRB = OFF_AGGR + (size_t)512*192*4;
static constexpr size_t OFF_FEATA = OFF_AGGRB + (size_t)512*24*4;       // float[512*1536]
static constexpr size_t OFF_BPW1  = OFF_FEATA + (size_t)512*1536*4;     // ushort[786432]
static constexpr size_t OFF_H1PRE = OFF_BPW1 + (size_t)786432*2;
static constexpr size_t OFF_H1    = OFF_H1PRE + (size_t)512*512*4;
static constexpr size_t OFF_BPW2  = OFF_H1   + (size_t)512*512*4;       // ushort[131072]
static constexpr size_t OFF_FA    = OFF_BPW2 + (size_t)131072*2;
static constexpr size_t OFF_BPH   = OFF_FA   + (size_t)512*256*4;       // ushort[2048]

// ---------- WMMA types ----------
typedef __attribute__((ext_vector_type(16))) __bf16 v16bf;
typedef __attribute__((ext_vector_type(8)))  float  v8f;
union BF16x16 { v16bf v; unsigned short u[16]; unsigned int w[8]; uint4 q[2]; };

__device__ __forceinline__ unsigned short f2bf(float f) {
    unsigned u = __float_as_uint(f);
    u += 0x7fffu + ((u >> 16) & 1u);            // round-to-nearest-even
    return (unsigned short)(u >> 16);
}
// pack two f32 -> two bf16 in one dword: 2x round-add + v_perm_b32
__device__ __forceinline__ unsigned bfpack(float lo, float hi) {
    unsigned a = __float_as_uint(lo);
    unsigned b = __float_as_uint(hi);
    a += 0x7fffu + ((a >> 16) & 1u);
    b += 0x7fffu + ((b >> 16) & 1u);
    return __builtin_amdgcn_perm(b, a, 0x07060302u);   // {b[31:16], a[31:16]}
}
__device__ __forceinline__ float leaky(float x) { return x >= 0.f ? x : 0.01f * x; }

// one K=32 GEMM step: load A f32 (cvt in-reg via v_perm pack) + packed-B bf16, WMMA
__device__ __forceinline__ v8f gemm_step(const float* __restrict__ arow,
                                         const unsigned short* __restrict__ bbase,
                                         int kc, int kh, v8f acc)
{
    float4 a0 = *(const float4*)(arow + kc * 32);
    float4 a1 = *(const float4*)(arow + kc * 32 + 4);
    float4 a2 = *(const float4*)(arow + kc * 32 + 16);
    float4 a3 = *(const float4*)(arow + kc * 32 + 20);
    BF16x16 a;
    a.w[0] = bfpack(a0.x, a0.y); a.w[1] = bfpack(a0.z, a0.w);
    a.w[2] = bfpack(a1.x, a1.y); a.w[3] = bfpack(a1.z, a1.w);
    a.w[4] = bfpack(a2.x, a2.y); a.w[5] = bfpack(a2.z, a2.w);
    a.w[6] = bfpack(a3.x, a3.y); a.w[7] = bfpack(a3.z, a3.w);
    const unsigned short* bp = bbase + ((size_t)kc * 2 + kh) * 256;
    BF16x16 b;
    b.q[0] = *(const uint4*)(bp);
    b.q[1] = *(const uint4*)(bp + 8);
    return __builtin_amdgcn_wmma_f32_16x16x32_bf16(false, a.v, false, b.v,
                                                   (short)0, acc, false, false);
}

// ============================================================
// Generic bf16 WMMA GEMM: C[M x N] = A[M x K] * B, A f32 (cvt in-reg),
// B pre-packed as [ng][kTile][half][col16][k16] bf16. kTiles must be even.
// ============================================================
template<int COLM>
__global__ __launch_bounds__(256) void k_gemm_bf16(
                            const float* __restrict__ A, int lda,
                            const unsigned short* __restrict__ Bp,
                            float* __restrict__ C,
                            int mTiles, int NG, int kTiles,
                            int nValid, int ldc, int ostride)
{
    int wave = threadIdx.x >> 5, lane = threadIdx.x & 31;
    int gw = blockIdx.x * 8 + wave;
    if (gw >= mTiles * NG) return;
    int mt = gw / NG, ng = gw % NG;
    int m = lane & 15, kh = lane >> 4;
    int row = mt * 16 + m;
    const float* arow = A + (size_t)row * lda + kh * 8;
    const unsigned short* bbase = Bp + ((size_t)ng * kTiles * 2) * 256 + m * 16;
    v8f acc0 = {}, acc1 = {};
    for (int kc = 0; kc < kTiles; kc += 2) {          // two accumulators break the
        acc0 = gemm_step(arow, bbase, kc,     kh, acc0);  // WMMA->WMMA RAW chain
        acc1 = gemm_step(arow, bbase, kc + 1, kh, acc1);
    }
    int n = ng * 16 + m;
    if (n < nValid) {
#pragma unroll
        for (int r = 0; r < 8; ++r) {
            int rr = mt * 16 + r + kh * 8;
            float v = acc0[r] + acc1[r];
            if (COLM) C[(size_t)n * ostride + rr] = v;
            else      C[(size_t)rr * ldc + n]     = v;
        }
    }
}

// Pack f32 matrix W[K x Nv] (row stride ldw) into WMMA-B bf16 layout.
__global__ void k_pack_b(const float* __restrict__ Wm, int ldw, int K, int Nv,
                         int NG, int kTiles, unsigned short* __restrict__ Bp)
{
    int t = blockIdx.x * blockDim.x + threadIdx.x;
    if (t >= NG * kTiles * 512) return;
    int kk = t & 15, col = (t >> 4) & 15, half = (t >> 8) & 1;
    int rest = t >> 9;
    int kc = rest % kTiles, ng = rest / kTiles;
    int k = kc * 32 + half * 8 + (kk & 7) + (kk >> 3) * 16, n = ng * 16 + col;
    float v = (k < K && n < Nv) ? Wm[(size_t)k * ldw + n] : 0.f;
    Bp[t] = f2bf(v);
}

// ============================================================
// Pair conv 136->128 ch, 3x3 as implicit GEMM (bf16 WMMA).
// Each wave: 32 pixels x 16 channels (two 16x16 tiles sharing B loads).
// pzn: [514][514][160] bf16 (normalized, zero borders/pad chans)
// Bp:  [ng8][tap9][kc5][half2][16][16]
// ============================================================
__global__ __launch_bounds__(256) void k_conv_pair(
                            const unsigned short* __restrict__ pzn,
                            const unsigned short* __restrict__ Bp,
                            const float* __restrict__ bias,
                            float* __restrict__ outp)
{
    int wave = threadIdx.x >> 5, lane = threadIdx.x & 31;
    int tile = blockIdx.x;                 // 8192 = 512 rows * 16 col-tiles
    int i = tile >> 4, j0 = (tile & 15) << 5;
    int m = lane & 15, kh = lane >> 4;
    int x0 = j0 + m;
    v8f acc0 = {}, acc1 = {}, acc2 = {}, acc3 = {};
#pragma unroll
    for (int tap = 0; tap < 9; ++tap) {
        const int dy = tap / 3, dx = tap - dy * 3;
        const unsigned short* ar0 =
            pzn + ((size_t)(i + dy) * 514 + (x0 + dx)) * 160 + kh * 8;
        const unsigned short* ar1 = ar0 + (size_t)16 * 160;
        const unsigned short* brow =
            Bp + ((size_t)(wave * 9 + tap) * 5 * 2) * 256 + m * 16;
        __builtin_prefetch(ar0 + (size_t)160 * 514, 0, 1);
#pragma unroll
        for (int kc = 0; kc < 5; ++kc) {
            BF16x16 b, a0, a1;
            const unsigned short* bp = brow + ((size_t)kc * 2 + kh) * 256;
            b.q[0] = *(const uint4*)(bp);
            b.q[1] = *(const uint4*)(bp + 8);
            a0.q[0] = *(const uint4*)(ar0 + kc * 32);
            a0.q[1] = *(const uint4*)(ar0 + kc * 32 + 16);
            a1.q[0] = *(const uint4*)(ar1 + kc * 32);
            a1.q[1] = *(const uint4*)(ar1 + kc * 32 + 16);
            if (kc & 1) {
                acc1 = __builtin_amdgcn_wmma_f32_16x16x32_bf16(false, a0.v, false, b.v,
                                                               (short)0, acc1, false, false);
                acc3 = __builtin_amdgcn_wmma_f32_16x16x32_bf16(false, a1.v, false, b.v,
                                                               (short)0, acc3, false, false);
            } else {
                acc0 = __builtin_amdgcn_wmma_f32_16x16x32_bf16(false, a0.v, false, b.v,
                                                               (short)0, acc0, false, false);
                acc2 = __builtin_amdgcn_wmma_f32_16x16x32_bf16(false, a1.v, false, b.v,
                                                               (short)0, acc2, false, false);
            }
        }
    }
    int n = (wave << 4) + m;
    float bsv = bias[n];
#pragma unroll
    for (int r = 0; r < 8; ++r) {
        int j = j0 + r + kh * 8;
        outp[((size_t)i * 512 + j) * 128 + n]        = leaky(acc0[r] + acc1[r] + bsv);
        outp[((size_t)i * 512 + j + 16) * 128 + n]   = leaky(acc2[r] + acc3[r] + bsv);
    }
}

// ============================================================ small VALU kernels
__global__ void k_zero_acc(float* acc) {
    int t = blockIdx.x * blockDim.x + threadIdx.x;
    if (t < 320) acc[t] = 0.f;
}

__global__ void k_proj(const float* __restrict__ x, const float* __restrict__ Wq,
                       const float* __restrict__ Wk, const float* __restrict__ Wv,
                       float* __restrict__ proj)
{
    int t = blockIdx.x * blockDim.x + threadIdx.x;
    if (t >= 512 * 576) return;
    int i = t / 576, col = t - i * 576;
    const float* Wm = (col < 192) ? Wq : (col < 384 ? Wk : Wv);
    int c = col % 192;
    const float* xr = x + i * 256;
    float s = 0.f;
    for (int k = 0; k < 256; ++k) s += xr[k] * Wm[k * 192 + c];
    proj[t] = s;
}

__global__ void k_transform(const float* __restrict__ proj, const float* __restrict__ R,
                            const float* __restrict__ tvec, float* __restrict__ qs,
                            float* __restrict__ ks, float* __restrict__ vp)
{
    int t = blockIdx.x * blockDim.x + threadIdx.x;
    if (t >= 512 * 192) return;
    int i = t / 192, pt = t - i * 192;
    int g = pt / 64, loc = pt - g * 64;
    const float* pr = proj + (size_t)i * 576 + g * 192 + loc * 3;
    const float* Rm = R + i * 9;
    const float* tv = tvec + i * 3;
    float p0 = pr[0], p1 = pr[1], p2 = pr[2];
    float* dst = (g == 0 ? qs : (g == 1 ? ks : vp)) + (size_t)i * 192 + loc * 3;
#pragma unroll
    for (int c = 0; c < 3; ++c)
        dst[c] = Rm[c * 3] * p0 + Rm[c * 3 + 1] * p1 + Rm[c * 3 + 2] * p2 + tv[c];
}

__global__ void k_sums(const float* __restrict__ qs, const float* __restrict__ ks,
                       float* __restrict__ q2, float* __restrict__ k2)
{
    int t = blockIdx.x * blockDim.x + threadIdx.x;
    if (t >= 4096) return;
    const float* a = qs + (size_t)t * 24;
    const float* b = ks + (size_t)t * 24;
    float sa = 0.f, sb = 0.f;
    for (int u = 0; u < 24; ++u) { sa += a[u] * a[u]; sb += b[u] * b[u]; }
    q2[t] = sa; k2[t] = sb;
}

__global__ void k_ssd(const float* __restrict__ qs, const float* __restrict__ ks,
                      const float* __restrict__ q2, const float* __restrict__ k2,
                      const float* __restrict__ gamma, float* __restrict__ lg)
{
    int t = blockIdx.x * blockDim.x + threadIdx.x;
    if (t >= CELLS) return;
    int i = t >> 9, j = t & 511;
    const float* qr = qs + (size_t)i * 192;
    const float* kr = ks + (size_t)j * 192;
    for (int h = 0; h < 8; ++h) {
        float d = 0.f;
        for (int u = 0; u < 24; ++u) d += qr[h * 24 + u] * kr[h * 24 + u];
        float ssd = q2[i * 8 + h] + k2[j * 8 + h] - 2.f * d;
        lg[(size_t)(8 + h) * CELLS + t] = ssd * (-gamma[h] * 0.0833333333f);
    }
}

__global__ void k_stats16(const float* __restrict__ lg, float* __restrict__ st16)
{
    int c = blockIdx.x, tid = threadIdx.x;
    const float* p = lg + (size_t)c * CELLS;
    float s = 0.f, ss = 0.f;
    for (int idx = tid; idx < CELLS; idx += 256) { float v = p[idx]; s += v; ss += v * v; }
    __shared__ float rs[256], rq[256];
    rs[tid] = s; rq[tid] = ss; __syncthreads();
    for (int off = 128; off; off >>= 1) {
        if (tid < off) { rs[tid] += rs[tid + off]; rq[tid] += rq[tid + off]; }
        __syncthreads();
    }
    if (!tid) {
        float m = rs[0] * (1.f / 262144.f);
        float var = rq[0] * (1.f / 262144.f) - m * m;
        st16[c * 2] = m; st16[c * 2 + 1] = rsqrtf(var + 1e-5f);
    }
}

__global__ void k_conv_att(const float* __restrict__ lg, const float* __restrict__ st16,
                           const float* __restrict__ caw, const float* __restrict__ cab,
                           const unsigned char* __restrict__ mask, float* __restrict__ lsb)
{
    __shared__ float sw[1152], sst[32], sb[8];
    int tid = threadIdx.x;
    for (int t = tid; t < 1152; t += 256) sw[t] = caw[t];
    if (tid < 32) sst[tid] = st16[tid];
    if (tid < 8)  sb[tid] = cab[tid];
    __syncthreads();
    int cell = blockIdx.x * 256 + tid;
    int i = cell >> 9, j = cell & 511;
    float acc[8] = {0.f,0.f,0.f,0.f,0.f,0.f,0.f,0.f};
    for (int ci = 0; ci < 16; ++ci) {
        float mn = sst[ci * 2], rs = sst[ci * 2 + 1];
        const float* ch = lg + (size_t)ci * CELLS;
        for (int ty = 0; ty < 3; ++ty) {
            int ii = i + ty - 1;
            for (int tx = 0; tx < 3; ++tx) {
                int jj = j + tx - 1;
                float v = 0.f;
                if ((unsigned)ii < 512u && (unsigned)jj < 512u)
                    v = (ch[(size_t)ii * 512 + jj] - mn) * rs;
                int tap = ty * 3 + tx;
#pragma unroll
                for (int h = 0; h < 8; ++h) acc[h] += sw[(h * 16 + ci) * 9 + tap] * v;
            }
        }
    }
    bool mij = mask[i] && mask[j];
#pragma unroll
    for (int h = 0; h < 8; ++h) {
        float v = leaky(acc[h] + sb[h]);
        if (!mij) v -= 1000000.f;
        lsb[((size_t)i * 8 + h) * 512 + j] = v;
    }
}

__global__ void k_softmax(const float* __restrict__ lsb, const unsigned char* __restrict__ mask,
                          float* __restrict__ alphaA)
{
    int b = blockIdx.x, tid = threadIdx.x;
    int i = b >> 3;
    const float* row = lsb + (size_t)b * 512;
    float v0 = row[tid], v1 = row[tid + 256];
    __shared__ float red[256];
    red[tid] = fmaxf(v0, v1); __syncthreads();
    for (int off = 128; off; off >>= 1) {
        if (tid < off) red[tid] = fmaxf(red[tid], red[tid + off]);
        __syncthreads();
    }
    float mx = red[0]; __syncthreads();
    float e0 = __expf(v0 - mx), e1 = __expf(v1 - mx);
    red[tid] = e0 + e1; __syncthreads();
    for (int off = 128; off; off >>= 1) {
        if (tid < off) red[tid] += red[tid + off];
        __syncthreads();
    }
    float inv = mask[i] ? (1.f / red[0]) : 0.f;
    alphaA[(size_t)b * 512 + tid] = e0 * inv;
    alphaA[(size_t)b * 512 + tid + 256] = e1 * inv;
}

__global__ void k_stats_z(const float* __restrict__ z, float* __restrict__ acc136)
{
    __shared__ float ssum[128], ssq[128];
    int tid = threadIdx.x;
    if (tid < 128) { ssum[tid] = 0.f; ssq[tid] = 0.f; }
    __syncthreads();
    size_t base = (size_t)blockIdx.x * 16384;
    for (int t = tid; t < 16384; t += 256) {
        size_t idx = base + t;
        float v = z[idx];
        int c = (int)(idx & 127);
        atomicAdd(&ssum[c], v);
        atomicAdd(&ssq[c], v * v);
    }
    __syncthreads();
    if (tid < 128) {
        atomicAdd(&acc136[tid], ssum[tid]);
        atomicAdd(&acc136[136 + tid], ssq[tid]);
    }
}

__global__ void k_stats_alpha(const float* __restrict__ alphaA, float* __restrict__ acc136)
{
    __shared__ float ssum[8], ssq[8];
    int tid = threadIdx.x;
    if (tid < 8) { ssum[tid] = 0.f; ssq[tid] = 0.f; }
    __syncthreads();
    size_t base = (size_t)blockIdx.x * 16384;
    for (int t = tid; t < 16384; t += 256) {
        size_t idx = base + t;
        float v = alphaA[idx];
        int c = (int)((idx >> 9) & 7);
        atomicAdd(&ssum[c], v);
        atomicAdd(&ssq[c], v * v);
    }
    __syncthreads();
    if (tid < 8) {
        atomicAdd(&acc136[128 + tid], ssum[tid]);
        atomicAdd(&acc136[136 + 128 + tid], ssq[tid]);
    }
}

__global__ void k_fin136(const float* __restrict__ acc136, float* __restrict__ st136)
{
    int t = blockIdx.x * blockDim.x + threadIdx.x;
    if (t >= 136) return;
    float m = acc136[t] * (1.f / 262144.f);
    float var = acc136[136 + t] * (1.f / 262144.f) - m * m;
    st136[t * 2] = m;
    st136[t * 2 + 1] = rsqrtf(var + 1e-5f);
}

__global__ void k_pack_pz(const float* __restrict__ z, const float* __restrict__ alphaA,
                          const float* __restrict__ st136, unsigned short* __restrict__ pzn)
{
    int cell = blockIdx.x, c = threadIdx.x;     // blockDim 160
    int i = cell >> 9, j = cell & 511;
    unsigned short o = 0;
    if (c < 136) {
        float v = (c < 128) ? z[(size_t)cell * 128 + c]
                            : alphaA[((size_t)i * 8 + (c - 128)) * 512 + j];
        o = f2bf((v - st136[c * 2]) * st136[c * 2 + 1]);
    }
    pzn[((size_t)(i + 1) * 514 + (j + 1)) * 160 + c] = o;
}

__global__ void k_border(unsigned short* __restrict__ pzn)
{
    int t = blockIdx.x * blockDim.x + threadIdx.x;
    if (t >= 4 * 514) return;
    int side = t / 514, pos = t - side * 514;
    int y, x;
    if (side == 0) { y = 0;   x = pos; }
    else if (side == 1) { y = 513; x = pos; }
    else if (side == 2) { y = pos; x = 0; }
    else { y = pos; x = 513; }
    uint4 zz = {0u, 0u, 0u, 0u};
    uint4* p = (uint4*)(pzn + ((size_t)y * 514 + x) * 160);
    for (int c = 0; c < 20; ++c) p[c] = zz;
}

__global__ void k_pack_cpw(const float* __restrict__ cpw, unsigned short* __restrict__ Bp)
{
    int t = blockIdx.x * blockDim.x + threadIdx.x;
    if (t >= 184320) return;
    int kk = t & 15, col = (t >> 4) & 15, half = (t >> 8) & 1;
    int rest = t >> 9;
    int kc = rest % 5, q = rest / 5;
    int tap = q % 9, ng = q / 9;
    int ci = kc * 32 + half * 8 + (kk & 7) + (kk >> 3) * 16;
    int c = ng * 16 + col;
    float v = 0.f;
    if (ci < 136) v = cpw[(((size_t)c * 136 + ci) * 3 + tap / 3) * 3 + (tap % 3)];
    Bp[t] = f2bf(v);
}

__global__ void k_feat_p2n(const float* __restrict__ z, const float* __restrict__ alphaA,
                           float* __restrict__ featA)
{
    int i = blockIdx.x, tid = threadIdx.x;      // 128 threads, tid = p
    __shared__ float sA[8][32];
    float acc[8] = {0.f,0.f,0.f,0.f,0.f,0.f,0.f,0.f};
    for (int jb = 0; jb < 16; ++jb) {
        for (int t = tid; t < 256; t += 128) {
            int h = t >> 5, jc = t & 31;
            sA[h][jc] = alphaA[((size_t)i * 8 + h) * 512 + jb * 32 + jc];
        }
        __syncthreads();
        for (int jc = 0; jc < 32; ++jc) {
            float zv = z[((size_t)i * 512 + jb * 32 + jc) * 128 + tid];
#pragma unroll
            for (int h = 0; h < 8; ++h) acc[h] += sA[h][jc] * zv;
        }
        __syncthreads();
    }
    float* frow = featA + (size_t)i * KPAD1;
#pragma unroll
    for (int h = 0; h < 8; ++h) frow[h * 128 + tid] = acc[h];
}

__global__ void k_aggr(const float* __restrict__ alphaA, const float* __restrict__ vpts,
                       const float* __restrict__ pcb, float* __restrict__ aggr,
                       float* __restrict__ aggrb)
{
    __shared__ float sA[4096];
    int i = blockIdx.x, tid = threadIdx.x;      // 256
    for (int t = tid; t < 4096; t += 256) sA[t] = alphaA[(size_t)i * 4096 + t];
    __syncthreads();
    if (tid < 192) {
        int h = tid / 24, rem = tid - h * 24;
        float s = 0.f;
        for (int j = 0; j < 512; ++j)
            s += sA[h * 512 + j] * vpts[(size_t)j * 192 + h * 24 + rem];
        aggr[(size_t)i * 192 + h * 24 + rem] = s;
    } else if (tid < 216) {
        int r = tid - 192, h = r / 3, c = r - h * 3;
        float s = 0.f;
        for (int j = 0; j < 512; ++j) s += sA[h * 512 + j] * pcb[j * 3 + c];
        aggrb[(size_t)i * 24 + h * 3 + c] = s;
    }
}

__global__ void k_feat2(const float* __restrict__ aggr, const float* __restrict__ aggrb,
                        const float* __restrict__ R, const float* __restrict__ tvec,
                        float* __restrict__ featA)
{
    int i = blockIdx.x, tid = threadIdx.x;      // 128
    __shared__ float Rm[9], tv[3];
    if (tid < 9) Rm[tid] = R[i * 9 + tid];
    if (tid < 3) tv[tid] = tvec[i * 3 + tid];
    __syncthreads();
    float* frow = featA + (size_t)i * KPAD1;
    if (tid < 64) {
        const float* a = aggr + (size_t)i * 192 + tid * 3;
        float d0 = a[0] - tv[0], d1 = a[1] - tv[1], d2 = a[2] - tv[2];
        float p0 = Rm[0] * d0 + Rm[3] * d1 + Rm[6] * d2;
        float p1 = Rm[1] * d0 + Rm[4] * d1 + Rm[7] * d2;
        float p2 = Rm[2] * d0 + Rm[5] * d1 + Rm[8] * d2;
        float fn = sqrtf(p0 * p0 + p1 * p1 + p2 * p2);
        float inv = 1.f / (fn + 1e-6f);
        frow[1024 + tid * 3 + 0] = p0;
        frow[1024 + tid * 3 + 1] = p1;
        frow[1024 + tid * 3 + 2] = p2;
        frow[1216 + tid] = fn;
        frow[1280 + tid * 3 + 0] = p0 * inv;
        frow[1280 + tid * 3 + 1] = p1 * inv;
        frow[1280 + tid * 3 + 2] = p2 * inv;
    } else if (tid < 72) {
        int h = tid - 64;
        const float* a = aggrb + (size_t)i * 24 + h * 3;
        float d0 = a[0] - tv[0], d1 = a[1] - tv[1], d2 = a[2] - tv[2];
        float p0 = Rm[0] * d0 + Rm[3] * d1 + Rm[6] * d2;
        float p1 = Rm[1] * d0 + Rm[4] * d1 + Rm[7] * d2;
        float p2 = Rm[2] * d0 + Rm[5] * d1 + Rm[8] * d2;
        float fn = sqrtf(p0 * p0 + p1 * p1 + p2 * p2);
        float inv = 1.f / (fn + 1e-6f);
        frow[1472 + h * 3 + 0] = p0;
        frow[1472 + h * 3 + 1] = p1;
        frow[1472 + h * 3 + 2] = p2;
        frow[1496 + h] = fn;
        frow[1504 + h * 3 + 0] = p0 * inv;
        frow[1504 + h * 3 + 1] = p1 * inv;
        frow[1504 + h * 3 + 2] = p2 * inv;
    } else if (tid < 80) {
        frow[1528 + (tid - 72)] = 0.f;          // zero K padding
    }
}

__global__ void k_ln1(const float* __restrict__ h1pre, const float* __restrict__ b1,
                      const float* __restrict__ g, const float* __restrict__ b,
                      float* __restrict__ h1)
{
    int i = blockIdx.x, tid = threadIdx.x;      // 256, cols 512
    const float* row = h1pre + (size_t)i * 512;
    float v0 = row[tid] + b1[tid];
    float v1 = row[tid + 256] + b1[tid + 256];
    __shared__ float rs[256], rq[256];
    rs[tid] = v0 + v1; rq[tid] = v0 * v0 + v1 * v1; __syncthreads();
    for (int off = 128; off; off >>= 1) {
        if (tid < off) { rs[tid] += rs[tid + off]; rq[tid] += rq[tid + off]; }
        __syncthreads();
    }
    float m = rs[0] * (1.f / 512.f);
    float var = rq[0] * (1.f / 512.f) - m * m;
    float rstd = rsqrtf(var + 1e-5f);
    h1[(size_t)i * 512 + tid]       = leaky((v0 - m) * rstd * g[tid] + b[tid]);
    h1[(size_t)i * 512 + tid + 256] = leaky((v1 - m) * rstd * g[tid + 256] + b[tid + 256]);
}

__global__ void k_out(const float* __restrict__ fa, const float* __restrict__ b2,
                      const float* __restrict__ x, const unsigned char* __restrict__ mask,
                      const float* __restrict__ lng, const float* __restrict__ lnb,
                      float* __restrict__ xout)
{
    int i = blockIdx.x, tid = threadIdx.x;      // 256
    float v = fa[(size_t)i * 256 + tid] + b2[tid];
    if (!mask[i]) v = 0.f;
    float r = x[(size_t)i * 256 + tid] + v;
    __shared__ float rs[256], rq[256];
    rs[tid] = r; rq[tid] = r * r; __syncthreads();
    for (int off = 128; off; off >>= 1) {
        if (tid < off) { rs[tid] += rs[tid + off]; rq[tid] += rq[tid + off]; }
        __syncthreads();
    }
    float m = rs[0] * (1.f / 256.f);
    float var = rq[0] * (1.f / 256.f) - m * m;
    float rstd = rsqrtf(var + 1e-5f);
    xout[(size_t)i * 256 + tid] = (r - m) * rstd * lng[tid] + lnb[tid];
}

// ============================================================ launch
extern "C" void kernel_launch(void* const* d_in, const int* in_sizes, int n_in,
                              void* d_out, int out_size, void* d_ws, size_t ws_size,
                              hipStream_t stream)
{
    const float* R    = (const float*)d_in[0];
    const float* tv   = (const float*)d_in[1];
    const float* pCB  = (const float*)d_in[2];
    const float* x    = (const float*)d_in[3];
    const float* z    = (const float*)d_in[4];
    const float* Wp2h = (const float*)d_in[5];
    const float* gamma= (const float*)d_in[6];
    const float* Wq   = (const float*)d_in[7];
    const float* Wk   = (const float*)d_in[8];
    const float* Wv   = (const float*)d_in[9];
    const float* W1   = (const float*)d_in[10];
    const float* b1   = (const float*)d_in[11];
    const float* ln1g = (const float*)d_in[12];
    const float* ln1b = (const float*)d_in[13];
    const float* W2   = (const float*)d_in[14];
    const float* b2   = (const float*)d_in[15];
    const float* caw  = (const float*)d_in[16];
    const float* cab  = (const float*)d_in[17];
    const float* lng  = (const float*)d_in[18];
    const float* lnb  = (const float*)d_in[19];
    const float* cpw  = (const float*)d_in[20];
    const float* cpb  = (const float*)d_in[21];
    const unsigned char* mask = (const unsigned char*)d_in[22];

    char* W = (char*)d_ws;
    float*          proj  = (float*)(W + OFF_PROJ);
    float*          lg    = (float*)(W + OFF_LG);
    float*          lsb   = (float*)(W + OFF_LS);
    unsigned short* pzn   = (unsigned short*)(W + OFF_PZN);
    float*          qs    = (float*)(W + OFF_QS);
    float*          ks    = (float*)(W + OFF_KS);
    float*          vpts  = (float*)(W + OFF_VPTS);
    float*          q2    = (float*)(W + OFF_Q2);
    float*          k2    = (float*)(W + OFF_K2);
    float*          st16  = (float*)(W + OFF_ST16);
    float*          alphA = (float*)(W + OFF_ALPH);
    float*          ac136 = (float*)(W + OFF_AC136);
    float*          st136 = (float*)(W + OFF_ST136);
    unsigned short* bpcp  = (unsigned short*)(W + OFF_BPCP);
    float*          aggr  = (float*)(W + OFF_AGGR);
    float*          aggrb = (float*)(W + OFF_AGGRB);
    float*          featA = (float*)(W + OFF_FEATA);
    unsigned short* bpw1  = (unsigned short*)(W + OFF_BPW1);
    float*          h1pre = (float*)(W + OFF_H1PRE);
    float*          h1    = (float*)(W + OFF_H1);
    unsigned short* bpw2  = (unsigned short*)(W + OFF_BPW2);
    float*          fa    = (float*)(W + OFF_FA);
    unsigned short* bph   = (unsigned short*)(W + OFF_BPH);

    float* xout = (float*)d_out;
    float* pout = xout + 512 * 256;

    // 0: zero stats accumulators (ws not re-poisoned between replays)
    k_zero_acc<<<2, 256, 0, stream>>>(ac136);
    // 1-3: projections + rigid transforms + squared sums
    k_proj<<<1152, 256, 0, stream>>>(x, Wq, Wk, Wv, proj);
    k_transform<<<384, 256, 0, stream>>>(proj, R, tv, qs, ks, vpts);
    k_sums<<<16, 256, 0, stream>>>(qs, ks, q2, k2);
    // 4: logits_pair = z @ W_p2h via WMMA (M=262144, K=128, N=16 pad)
    k_pack_b<<<8, 256, 0, stream>>>(Wp2h, 8, 128, 8, 1, 4, bph);
    k_gemm_bf16<1><<<2048, 256, 0, stream>>>(z, 128, bph, lg,
                                             16384, 1, 4, 8, 0, CELLS);
    // 5: spatial logits
    k_ssd<<<1024, 256, 0, stream>>>(qs, ks, q2, k2, gamma, lg);
    // 6: instance norm stats (16 ch) + attention conv + mask
    k_stats16<<<16, 256, 0, stream>>>(lg, st16);
    k_conv_att<<<1024, 256, 0, stream>>>(lg, st16, caw, cab, mask, lsb);
    // 7: softmax over j per (i,h)
    k_softmax<<<4096, 256, 0, stream>>>(lsb, mask, alphA);
    // 8: stats for pz instance norm (z 128 ch + alpha 8 ch)
    k_stats_z<<<2048, 256, 0, stream>>>(z, ac136);
    k_stats_alpha<<<128, 256, 0, stream>>>(alphA, ac136);
    k_fin136<<<1, 256, 0, stream>>>(ac136, st136);
    // 9: pack normalized pz into padded bf16 [514][514][160]
    k_pack_pz<<<CELLS, 160, 0, stream>>>(z, alphA, st136, pzn);
    k_border<<<9, 256, 0, stream>>>(pzn);
    // 10: pair conv via WMMA implicit GEMM -> pair_out (dominant kernel)
    k_pack_cpw<<<720, 256, 0, stream>>>(cpw, bpcp);
    k_conv_pair<<<8192, 256, 0, stream>>>(pzn, bpcp, cpb, pout);
    // 11: feature aggregation -> feat_all (padded to 1536)
    k_feat_p2n<<<512, 128, 0, stream>>>(z, alphA, featA);
    k_aggr<<<512, 256, 0, stream>>>(alphA, vpts, pCB, aggr, aggrb);
    k_feat2<<<512, 128, 0, stream>>>(aggr, aggrb, R, tv, featA);
    // 12: FFN via WMMA + LayerNorms
    k_pack_b<<<3072, 256, 0, stream>>>(W1, 512, 1528, 512, 32, 48, bpw1);
    k_gemm_bf16<0><<<128, 256, 0, stream>>>(featA, KPAD1, bpw1, h1pre,
                                            32, 32, 48, 512, 512, 0);
    k_ln1<<<512, 256, 0, stream>>>(h1pre, b1, ln1g, ln1b, h1);
    k_pack_b<<<512, 256, 0, stream>>>(W2, 256, 512, 256, 16, 16, bpw2);
    k_gemm_bf16<0><<<64, 256, 0, stream>>>(h1, 512, bpw2, fa,
                                           32, 16, 16, 256, 256, 0);
    k_out<<<512, 256, 0, stream>>>(fa, b2, x, mask, lng, lnb, xout);
}
